// TrueReservoir_58213986730249
// MI455X (gfx1250) — matrix operand order repaired and checked by reading, not compiled, not run on hardware
//
#include <hip/hip_runtime.h>
#include <hip/hip_bf16.h>

typedef __bf16 bf16;
typedef __attribute__((ext_vector_type(16))) __bf16 v16bf;
typedef __attribute__((ext_vector_type(8)))  __bf16 v8bf;
typedef __attribute__((ext_vector_type(8)))  float  v8f;

#define RB 32
#define RS 2048
#define RI 512
#define RH 1024
#define LEAK 0.95f
#define PAD 8   // bf16 elements of row padding in LDS (16B, keeps alignment)

// ---- raw LDS byte offset of a pointer into __shared__ ----
__device__ __forceinline__ unsigned lds_offset(const void* p) {
  return (unsigned)(unsigned long long)(const __attribute__((address_space(3))) char*)p;
}

// ---- CDNA5 async copy: 16 bytes global -> LDS per lane (ASYNCcnt tracked) ----
__device__ __forceinline__ void async_copy16(unsigned lds_off, const void* gptr) {
  unsigned long long g = (unsigned long long)gptr;
  asm volatile("global_load_async_to_lds_b128 %0, %1, off"
               :: "v"(lds_off), "v"(g) : "memory");
}
__device__ __forceinline__ void wait_async() {
  asm volatile("s_wait_asynccnt 0x0" ::: "memory");
}

// ---- WMMA fragment loaders from GLOBAL (layouts per CDNA5 ISA §7.12.2) ----
// A: 16x32 bf16 row-major tile, base = &A[m0,k0], ld = row stride.
__device__ __forceinline__ v16bf load_a_frag(const bf16* __restrict__ base, int ld, int lane) {
  const int m  = lane & 15;
  const int kh = lane >> 4;
  const bf16* p = base + (size_t)m * ld + kh * 8;
  v8bf lo = *(const v8bf*)(p);
  v8bf hi = *(const v8bf*)(p + 16);
  v16bf a;
#pragma unroll
  for (int i = 0; i < 8; ++i) { a[i] = lo[i]; a[i + 8] = hi[i]; }
  return a;
}
// B: 32x16 with B[k,n] = W[n0+n, k0+k], W row-major.
__device__ __forceinline__ v16bf load_b_frag(const bf16* __restrict__ base, int ld, int lane) {
  const int n  = lane & 15;
  const int kh = lane >> 4;
  const bf16* p = base + (size_t)n * ld + kh * 8;
  v8bf lo = *(const v8bf*)(p);
  v8bf hi = *(const v8bf*)(p + 16);
  v16bf b;
#pragma unroll
  for (int i = 0; i < 8; ++i) { b[i] = lo[i]; b[i + 8] = hi[i]; }
  return b;
}

// ---- fp32 -> bf16 conversion ----
__global__ void cvt_kernel(const float* __restrict__ s, bf16* __restrict__ d, int n) {
  int i = blockIdx.x * blockDim.x + threadIdx.x;
  int stride = blockDim.x * gridDim.x;
  for (; i < n; i += stride) d[i] = (bf16)s[i];
}

// ---- init: zero state double-buffer + barrier control words ----
__global__ void init_kernel(bf16* __restrict__ st, unsigned* __restrict__ ctrl) {
  int i = blockIdx.x * blockDim.x + threadIdx.x;
  if (i < 64) ctrl[i] = 0u;
  int stride = blockDim.x * gridDim.x;
  for (int j = i; j < 2 * RB * RH; j += stride) st[j] = (bf16)0.0f;
}

// ---- input projection: U[r,h] = sum_i Xb[r,i] * Winb[h,i] ----
// One wave computes a 16x64 strip (4 N-tiles, A fragment reused 4x). Memory-bound;
// ~320 MB total traffic => ~15 us at 23.3 TB/s, WMMA throughput irrelevant here.
__global__ void proj_kernel(const bf16* __restrict__ X,    // [R=RB*RS, RI]
                            const bf16* __restrict__ Win,  // [RH, RI]
                            float* __restrict__ U) {       // [R, RH]
  const int lane = threadIdx.x & 31;
  const int task = blockIdx.x * (blockDim.x >> 5) + (threadIdx.x >> 5);
  const int mt = task >> 4;
  const int ng = task & 15;
  const int m0 = mt << 4;
  const int n0 = ng << 6;
  v8f acc[4] = {};
  for (int k0 = 0; k0 < RI; k0 += 32) {
    v16bf a = load_a_frag(X + (size_t)m0 * RI + k0, RI, lane);
#pragma unroll
    for (int j = 0; j < 4; ++j) {
      v16bf b = load_b_frag(Win + (size_t)(n0 + 16 * j) * RI + k0, RI, lane);
      acc[j] = __builtin_amdgcn_wmma_f32_16x16x32_bf16(
          false, a, false, b, (short)0, acc[j], false, false);
    }
  }
  const int kh = lane >> 4, nn = lane & 15;
#pragma unroll
  for (int j = 0; j < 4; ++j) {
#pragma unroll
    for (int v = 0; v < 8; ++v) {
      int r = m0 + v + 8 * kh;
      int c = n0 + 16 * j + nn;
      U[(size_t)r * RH + c] = acc[j][v];
    }
  }
}

// ---- device-scope split barrier (per batch-half), graph-capture safe ----
__device__ __forceinline__ void group_barrier(unsigned* cnt, unsigned* gen, unsigned nblk) {
  __syncthreads();
  if (threadIdx.x == 0) {
    unsigned g = __hip_atomic_load(gen, __ATOMIC_RELAXED, __HIP_MEMORY_SCOPE_AGENT);
    unsigned arrived = __hip_atomic_fetch_add(cnt, 1u, __ATOMIC_ACQ_REL, __HIP_MEMORY_SCOPE_AGENT);
    if (arrived == nblk - 1u) {
      __hip_atomic_store(cnt, 0u, __ATOMIC_RELEASE, __HIP_MEMORY_SCOPE_AGENT);
      __hip_atomic_fetch_add(gen, 1u, __ATOMIC_ACQ_REL, __HIP_MEMORY_SCOPE_AGENT);
    } else {
      while (__hip_atomic_load(gen, __ATOMIC_ACQUIRE, __HIP_MEMORY_SCOPE_AGENT) == g)
        __builtin_amdgcn_s_sleep(1);
    }
  }
  __syncthreads();
}

// ---- persistent recurrent scan ----
// grid = 64 blocks x 64 threads (2 waves). block b: mt = b>>5 (batch half),
// waves cover n-tiles (b&31)*2 .. +1.  Per step:
//   * state tile (16x1024 bf16, 32KB) async-staged into LDS (one round trip)
//   * W_res panels live in LDS for the whole kernel (loaded once)
//   * 32 chained WMMAs read both fragments from LDS (ds_load_b128)
//   * leak term 'prev' carried in VGPRs (no global read-back on critical path)
__global__ void __launch_bounds__(64)
scan_kernel(const float* __restrict__ U,     // [RB*RS, RH]
            const bf16* __restrict__ Wres,   // [RH, RH] bf16 (L2-resident)
            bf16* state,                     // 2 * RB*RH (double buffer)
            float* __restrict__ out,         // [RB, RS, RH] (write-only, NT)
            unsigned* __restrict__ ctrl) {
  __shared__ bf16 sW[2][16][RH + PAD];   // per-wave W_res panel (W[n0+n, k])
  __shared__ bf16 sA[16][RH + PAD];      // current state tile (shared by both waves)

  const int lane = threadIdx.x & 31;
  const int wav  = threadIdx.x >> 5;   // 0..1
  const int mt   = blockIdx.x >> 5;    // 0..1 (batch half)
  const int nb   = blockIdx.x & 31;
  const int m0   = mt << 4;
  const int n0   = ((nb << 1) + wav) << 4;
  unsigned* cnt = ctrl + mt;
  unsigned* gen = ctrl + 32 + mt;
  const int kh = lane >> 4, nn = lane & 15;
  const int h  = n0 + nn;

  // ---- prologue: stage this wave's W_res panel into LDS (time-invariant) ----
  for (int s = lane; s < 16 * 64; s += 32) {          // 1024 x 16B segments
    int row = s >> 6, col = (s & 63) << 3;
    async_copy16(lds_offset(&sW[wav][row][col]),
                 Wres + (size_t)(n0 + row) * RH + col);
  }
  wait_async();
  __syncthreads();

  float prev[8];
#pragma unroll
  for (int v = 0; v < 8; ++v) prev[v] = 0.0f;

  for (int t = 0; t < RS; ++t) {
    const bf16* st = state + (size_t)(t & 1) * (RB * RH);
    bf16* stn      = state + (size_t)((t + 1) & 1) * (RB * RH);

    // async-stage the state tile for this step (both waves cooperate)
    for (int s = threadIdx.x; s < 16 * 64; s += 64) { // 1024 x 16B segments
      int row = s >> 6, col = (s & 63) << 3;
      async_copy16(lds_offset(&sA[row][col]),
                   st + (size_t)(m0 + row) * RH + col);
    }

    // overlap the async copies with the u_t tile fetch (read-once -> NT)
    v8f acc;
#pragma unroll
    for (int v = 0; v < 8; ++v) {
      int b = m0 + v + 8 * kh;
      acc[v] = __builtin_nontemporal_load(&U[((size_t)b * RS + t) * RH + h]);
    }
    if (t + 1 < RS)
      __builtin_prefetch(&U[((size_t)(m0 + 8 * kh) * RS + (t + 1)) * RH + h], 0, 1);

    wait_async();
    __syncthreads();

    // acc += state(16x1024) x Wres^T(1024x16): 32 chained WMMAs, frags from LDS
#pragma unroll 4
    for (int k0 = 0; k0 < RH; k0 += 32) {
      const bf16* pa = &sA[lane & 15][k0 + (kh << 3)];
      v8bf alo = *(const v8bf*)pa;
      v8bf ahi = *(const v8bf*)(pa + 16);
      const bf16* pb = &sW[wav][lane & 15][k0 + (kh << 3)];
      v8bf blo = *(const v8bf*)pb;
      v8bf bhi = *(const v8bf*)(pb + 16);
      v16bf a, bm;
#pragma unroll
      for (int i = 0; i < 8; ++i) {
        a[i] = alo[i];  a[i + 8] = ahi[i];
        bm[i] = blo[i]; bm[i + 8] = bhi[i];
      }
      acc = __builtin_amdgcn_wmma_f32_16x16x32_bf16(
          false, a, false, bm, (short)0, acc, false, false);
    }

    // tanh + leaky update; prev stays in VGPRs; out is streamed non-temporal
#pragma unroll
    for (int v = 0; v < 8; ++v) {
      int b = m0 + v + 8 * kh;
      float nv = LEAK * tanhf(acc[v]) + (1.0f - LEAK) * prev[v];
      prev[v] = nv;
      __builtin_nontemporal_store(nv, &out[((size_t)b * RS + t) * RH + h]);
      stn[(size_t)b * RH + h] = (bf16)nv;   // regular-temporal: re-read next step
    }

    group_barrier(cnt, gen, 32u);  // sync this batch half before next step
  }
}

extern "C" void kernel_launch(void* const* d_in, const int* in_sizes, int n_in,
                              void* d_out, int out_size, void* d_ws, size_t ws_size,
                              hipStream_t stream) {
  (void)in_sizes; (void)n_in; (void)out_size; (void)ws_size;
  const float* x    = (const float*)d_in[0];  // [RB, RS, RI]
  const float* Win  = (const float*)d_in[1];  // [RH, RI]
  const float* Wres = (const float*)d_in[2];  // [RH, RH]
  float* out = (float*)d_out;                 // [RB, RS, RH]

  char* ws = (char*)d_ws;
  size_t off = 0;
  auto take = [&](size_t bytes) -> char* {
    char* p = ws + off;
    off = (off + bytes + 255) & ~(size_t)255;
    return p;
  };
  unsigned* ctrl = (unsigned*)take(256);                     // cnt[0..1] @ +0, gen[0..1] @ +128B
  bf16*  xb    = (bf16*)take((size_t)RB * RS * RI * 2);      //  64 MiB
  bf16*  winb  = (bf16*)take((size_t)RH * RI * 2);           //   1 MiB
  bf16*  wresb = (bf16*)take((size_t)RH * RH * 2);           //   2 MiB (L2-resident)
  bf16*  st    = (bf16*)take((size_t)2 * RB * RH * 2);       // 128 KiB state double buffer
  float* U     = (float*)take((size_t)RB * RS * RH * 4);     // 256 MiB input drive

  hipLaunchKernelGGL(init_kernel, dim3(64),   dim3(256), 0, stream, st, ctrl);
  hipLaunchKernelGGL(cvt_kernel,  dim3(8192), dim3(256), 0, stream, x,    xb,    RB * RS * RI);
  hipLaunchKernelGGL(cvt_kernel,  dim3(512),  dim3(256), 0, stream, Win,  winb,  RH * RI);
  hipLaunchKernelGGL(cvt_kernel,  dim3(1024), dim3(256), 0, stream, Wres, wresb, RH * RH);
  hipLaunchKernelGGL(proj_kernel, dim3(8192), dim3(256), 0, stream, xb, winb, U);
  hipLaunchKernelGGL(scan_kernel, dim3(64),   dim3(64),  0, stream, U, wresb, st, out, ctrl);
}
